// baselineSNN_17308718203127
// MI455X (gfx1250) — matrix operand order repaired and checked by reading, not compiled, not run on hardware
//
#include <hip/hip_runtime.h>

typedef __attribute__((ext_vector_type(16))) _Float16 v16h;
typedef __attribute__((ext_vector_type(8)))  _Float16 h8;
typedef __attribute__((ext_vector_type(8)))  float    v8f;
typedef __attribute__((ext_vector_type(4)))  float    f4;

// ---------------- problem constants ----------------
constexpr int T    = 15;
constexpr int C1I  = 6,  C1O = 30;
constexpr int H1   = 256;                   // conv1 output res
constexpr int HP1  = 260;                   // padded input res (pad 2)
constexpr int C2O  = 250;
constexpr int H2   = 128;                   // conv2 output res
constexpr int HP2  = 130;                   // padded pooled res (pad 1)
constexpr int HW2  = H2 * H2;               // 16384
constexpr int NTOT = C2O * HW2;             // 4,096,000  (C,H,W)
constexpr long NOUT = (long)T * NTOT;       // 61,440,000 (T,C,H,W)

constexpr int KT1 = 5, KT2 = 9;             // K chunks of 32 (150->160, 270->288)
constexpr int NT1 = 2, NT2 = 16;            // N tiles of 16 (30->32, 250->256)

// ---------------- workspace layout ----------------
constexpr size_t alignup(size_t x) { return (x + 255) & ~(size_t)255; }
constexpr size_t W1P_OFF  = 0;
constexpr size_t W2P_OFF  = alignup(W1P_OFF  + (size_t)KT1 * NT1 * 32 * 16 * 2);   // 10,240
constexpr size_t XPAD_OFF = alignup(W2P_OFF  + (size_t)KT2 * NT2 * 32 * 16 * 2);   // 147,456
constexpr size_t PPAD_OFF = alignup(XPAD_OFF + (size_t)T * C1I * HP1 * HP1 * 2);   // 12.17 MB
constexpr size_t PA_OFF   = alignup(PPAD_OFF + (size_t)T * C1O * HP2 * HP2 * 2);   // 15.21 MB
constexpr size_t WIN_OFF  = alignup(PA_OFF   + (size_t)T * HW2 * 4);
constexpr size_t NSP_OFF  = alignup(WIN_OFF  + (size_t)HW2 * 4);
constexpr size_t VAL_OFF  = alignup(NSP_OFF  + (size_t)NTOT * 4);
constexpr size_t TOT_OFF  = alignup(VAL_OFF  + (size_t)NTOT * 4);
constexpr size_t VMAX_OFF = alignup(TOT_OFF  + (size_t)NTOT * 4);
constexpr size_t REDV_OFF = alignup(VMAX_OFF + 256);
constexpr size_t REDI_OFF = alignup(REDV_OFF + 4096);
constexpr size_t WB_OFF   = alignup(REDI_OFF + 4096);

// WMMA fragment index maps (CDNA5 ISA, wave32)
__host__ __device__ constexpr int kmapA(int e, int g) { return (e & 7) + ((e >> 3) << 4) + (g << 3); }
__host__ __device__ constexpr int kmapB(int e, int g) { return (e & 7) + ((e >> 3) << 4) + (g << 3); }

// compile-time element offsets into padded activation buffers (out-of-K -> 0, weight is 0 there)
__host__ __device__ constexpr int eoff1(int kk) {
    return (kk < 150) ? ((kk / 25) * (HP1 * HP1) + ((kk % 25) / 5) * HP1 + (kk % 5)) : 0;
}
__host__ __device__ constexpr int eoff2(int kk) {
    return (kk < 270) ? ((kk / 9) * (HP2 * HP2) + ((kk % 9) / 3) * HP2 + (kk % 3)) : 0;
}

// async copy of 16B granules into LDS (CDNA5 GLOBAL_LOAD_ASYNC_TO_LDS, GVS mode)
__device__ __forceinline__ void async_b128(unsigned lds_addr, unsigned goff,
                                           unsigned long long gbase) {
    asm volatile("global_load_async_to_lds_b128 %0, %1, %2"
                 :: "v"(lds_addr), "v"(goff), "s"(gbase) : "memory");
}
__device__ __forceinline__ void wait_async0() {
    asm volatile("s_wait_asynccnt 0x0" ::: "memory");
}

// ---------------- weight pre-swizzle into B-fragment order ----------------
__global__ void prep_w1(const float* __restrict__ w1, _Float16* __restrict__ w1p) {
    int idx = blockIdx.x * 256 + threadIdx.x;                 // [kt][nt][lane][e]
    if (idx >= KT1 * NT1 * 32 * 16) return;
    int e = idx & 15, lane = (idx >> 4) & 31, nt = (idx >> 9) & 1, kt = idx >> 10;
    int n  = nt * 16 + (lane & 15);
    int kk = kt * 32 + kmapB(e, lane >> 4);
    float v = 0.f;
    if (n < C1O && kk < 150) {
        int ci = kk / 25, kh = (kk % 25) / 5, kw = kk % 5;
        v = w1[((n * C1I + ci) * 5 + kh) * 5 + kw];
    }
    w1p[idx] = (_Float16)v;
}

__global__ void prep_w2(const float* __restrict__ w2, _Float16* __restrict__ w2p) {
    int idx = blockIdx.x * 256 + threadIdx.x;                 // [kt][nt][lane][e]
    if (idx >= KT2 * NT2 * 32 * 16) return;
    int e = idx & 15, lane = (idx >> 4) & 31, nt = (idx >> 9) & 15, kt = idx >> 13;
    int n  = nt * 16 + (lane & 15);
    int kk = kt * 32 + kmapB(e, lane >> 4);
    float v = 0.f;
    if (n < C2O && kk < 270) {
        int ci = kk / 9, kh = (kk % 9) / 3, kw = kk % 3;
        v = w2[((n * C1O + ci) * 3 + kh) * 3 + kw];
    }
    w2p[idx] = (_Float16)v;
}

// ---------------- padded fp16 input image ----------------
__global__ void xpad_fill(const float* __restrict__ x, _Float16* __restrict__ xpad) {
    int idx = blockIdx.x * 256 + threadIdx.x;                 // [t*6+ci][260][260]
    if (idx >= T * C1I * HP1 * HP1) return;
    int cp = idx % HP1, hp = (idx / HP1) % HP1, tc = idx / (HP1 * HP1);
    int h = hp - 2, c = cp - 2;
    float v = 0.f;
    if (h >= 0 && h < H1 && c >= 0 && c < H1) v = x[((size_t)tc * H1 + h) * H1 + c];
    xpad[idx] = (_Float16)v;
}

// ---------------- conv1: WMMA f16, async-LDS weights, LDS-transposed spike store ----------------
__global__ __launch_bounds__(256) void conv1_fire(const _Float16* __restrict__ xpad,
                                                  const _Float16* __restrict__ w1p,
                                                  _Float16* __restrict__ spk1) {
    __shared__ __align__(16) _Float16 w1s[KT1 * NT1 * 32 * 16];   // 10 KB weights
    __shared__ __align__(16) _Float16 tls[8][16 * 16];            // per-wave 512B transpose tile
    const int tid  = threadIdx.x;
    const int lane = tid & 31;
    const int wave = tid >> 5;
    const int tile = blockIdx.x * 8 + wave;                   // 61,440 M-tiles of 16 pixels
    const int t   = tile / (H1 * 16);
    const int rem = tile % (H1 * 16);
    const int r   = rem >> 4;
    const int c0  = (rem & 15) << 4;
    const int g   = lane >> 4, ln = lane & 15;
    const unsigned pixbase = (unsigned)(((t * C1I * HP1 + r) * HP1) + c0 + ln);

    {   // one-shot async stage of all conv1 weight fragments (640 x 16B)
        const unsigned long long gb = (unsigned long long)(const void*)w1p;
        unsigned lbase = (unsigned)(size_t)(&w1s[0]);
#pragma unroll
        for (int j = 0; j < 3; ++j) {
            int idx = j * 256 + tid;
            if (idx < 640) async_b128(lbase + idx * 16u, idx * 16u, gb);
        }
        wait_async0();
        __syncthreads();
    }

    v8f acc0 = {}; v8f acc1 = {};
    const v16h* w1v = (const v16h*)&w1s[0];

#pragma unroll
    for (int kt = 0; kt < KT1; ++kt) {
        v16h a;
#pragma unroll
        for (int e = 0; e < 16; ++e) {
            const int o0 = eoff1(kt * 32 + kmapA(e, 0));      // compile-time literals
            const int o1 = eoff1(kt * 32 + kmapA(e, 1));
            a[e] = xpad[pixbase + (unsigned)(g ? o1 : o0)];   // saddr + 32-bit voffset
        }
        v16h b0 = w1v[(kt * NT1 + 0) * 32 + lane];            // ds_load_b128 x2
        v16h b1 = w1v[(kt * NT1 + 1) * 32 + lane];
        acc0 = __builtin_amdgcn_wmma_f32_16x16x32_f16(false, a, false, b0, (short)0, acc0, false, false);
        acc1 = __builtin_amdgcn_wmma_f32_16x16x32_f16(false, a, false, b1, (short)0, acc1, false, false);
    }

    // fire + per-wave LDS transpose -> coalesced b128 stores (same-wave DS ops are in-order)
    const int ch  = lane >> 1;                                // reader lane mapping
    const int px0 = (lane & 1) * 8;
#pragma unroll
    for (int nt = 0; nt < NT1; ++nt) {
        h8 sp;
#pragma unroll
        for (int rr = 0; rr < 8; ++rr) {
            float p = (nt == 0) ? acc0[rr] : acc1[rr];
            sp[rr] = (_Float16)((p > 15.f) ? 1.f : 0.f);      // D: M = rr+8g, N = ln
        }
        *(h8*)&tls[wave][ln * 16 + g * 8] = sp;               // ds_store_b128
        h8 row = *(h8*)&tls[wave][ch * 16 + px0];             // ds_load_b128
        const int n = nt * 16 + ch;
        if (n < C1O)
            *(h8*)&spk1[((t * C1O + n) * H1 + r) * H1 + c0 + px0] = row;  // global_store_b128
    }
}

// ---------------- 2x2 maxpool of binary spikes -> padded pooled map ----------------
__global__ void pool_k(const _Float16* __restrict__ spk1, _Float16* __restrict__ ppad) {
    int idx = blockIdx.x * 256 + threadIdx.x;                 // [t*30+c][130][130]
    if (idx >= T * C1O * HP2 * HP2) return;
    int wp = idx % HP2, hp = (idx / HP2) % HP2, tc = idx / (HP2 * HP2);
    float m = 0.f;
    if (hp >= 1 && hp <= H2 && wp >= 1 && wp <= H2) {
        const _Float16* p = spk1 + (((size_t)tc * H1 + 2 * (hp - 1)) * H1 + 2 * (wp - 1));
        m = fmaxf(fmaxf((float)p[0], (float)p[1]), fmaxf((float)p[H1], (float)p[H1 + 1]));
    }
    ppad[idx] = (_Float16)m;
}

// ---------------- conv2: WMMA f16, async-LDS double-buffered weights, LDS-transposed store ----------------
__global__ __launch_bounds__(256) void conv2_fire(const _Float16* __restrict__ ppad,
                                                  const _Float16* __restrict__ w2p,
                                                  float* __restrict__ pot) {
    __shared__ __align__(32) _Float16 bsm[2][NT2 * 32 * 16];  // 2 x 16 KB weight slices
    __shared__ __align__(16) float    tls[8][16 * 16];        // per-wave 1KB transpose tile
    const int tid  = threadIdx.x;
    const int lane = tid & 31;
    const int wave = tid >> 5;
    const int tile = blockIdx.x * 8 + wave;                   // 15,360 M-tiles of 16 pixels
    const int t   = tile / (H2 * 8);
    const int rem = tile % (H2 * 8);
    const int r   = rem >> 3;
    const int c0  = (rem & 7) << 4;
    const int g   = lane >> 4, ln = lane & 15;
    const unsigned pixbase = (unsigned)(((t * C1O * HP2 + r) * HP2) + c0 + ln);
    const unsigned long long wgb = (unsigned long long)(const void*)w2p;

    auto stage = [&](int kt, int nb) {                        // 16 KB slice, 4 x b128 per thread
        unsigned lbase = (unsigned)(size_t)(&bsm[nb][0]) + (unsigned)tid * 16u;
        unsigned gbase = (unsigned)(kt * NT2 * 32 * 32) + (unsigned)tid * 16u;
#pragma unroll
        for (int j = 0; j < 4; ++j) async_b128(lbase + j * 4096u, gbase + j * 4096u, wgb);
    };

    v8f acc[NT2];
    v8f zero = {};
#pragma unroll
    for (int i = 0; i < NT2; ++i) acc[i] = zero;

    stage(0, 0);
#pragma unroll
    for (int kt = 0; kt < KT2; ++kt) {
        wait_async0();                                        // our async writes done
        __syncthreads();                                      // everyone's writes visible
        if (kt + 1 < KT2) stage(kt + 1, (kt + 1) & 1);        // prefetch next slice

        v16h a;
#pragma unroll
        for (int e = 0; e < 16; ++e) {
            const int o0 = eoff2(kt * 32 + kmapA(e, 0));      // compile-time literals
            const int o1 = eoff2(kt * 32 + kmapA(e, 1));
            a[e] = ppad[pixbase + (unsigned)(g ? o1 : o0)];
        }
        const v16h* bs = (const v16h*)&bsm[kt & 1][0];
#pragma unroll
        for (int nt = 0; nt < NT2; ++nt) {
            v16h b = bs[nt * 32 + lane];                      // ds_load_b128 x2
            acc[nt] = __builtin_amdgcn_wmma_f32_16x16x32_f16(false, a, false, b, (short)0, acc[nt], false, false);
        }
    }

    // fire + per-wave LDS transpose -> coalesced b128 stores of pot
    const int ch  = lane >> 1;
    const int px0 = (lane & 1) * 8;
#pragma unroll
    for (int nt = 0; nt < NT2; ++nt) {
        f4 lo = { (acc[nt][0] > 10.f) ? acc[nt][0] : 0.f, (acc[nt][1] > 10.f) ? acc[nt][1] : 0.f,
                  (acc[nt][2] > 10.f) ? acc[nt][2] : 0.f, (acc[nt][3] > 10.f) ? acc[nt][3] : 0.f };
        f4 hi = { (acc[nt][4] > 10.f) ? acc[nt][4] : 0.f, (acc[nt][5] > 10.f) ? acc[nt][5] : 0.f,
                  (acc[nt][6] > 10.f) ? acc[nt][6] : 0.f, (acc[nt][7] > 10.f) ? acc[nt][7] : 0.f };
        *(f4*)&tls[wave][ln * 16 + g * 8 + 0] = lo;           // ds_store_b128 x2
        *(f4*)&tls[wave][ln * 16 + g * 8 + 4] = hi;
        f4 r0 = *(f4*)&tls[wave][ch * 16 + px0 + 0];          // ds_load_b128 x2
        f4 r1 = *(f4*)&tls[wave][ch * 16 + px0 + 4];
        const int n = nt * 16 + ch;
        if (n < C2O) {
            float* dst = &pot[(((size_t)t * C2O + n) * H2 + r) * H2 + c0 + px0];
            *(f4*)(dst + 0) = r0;                             // global_store_b128 x2
            *(f4*)(dst + 4) = r1;
        }
    }
}

// ---------------- pointwise inhibition ----------------
__global__ void inhib_a1(const float* __restrict__ pot, int* __restrict__ pa) {
    int idx = blockIdx.x * 256 + threadIdx.x;                 // (t, h, w)
    if (idx >= T * HW2) return;
    int hw = idx & (HW2 - 1), t = idx >> 14;
    const float* p = pot + (size_t)t * NTOT + hw;
    float mv = 0.f; int mi = 0;
    for (int c = 0; c < C2O; ++c) {
        float v = p[(size_t)c * HW2];
        if (v > mv) { mv = v; mi = c; }                       // strict > => first-index ties
    }
    pa[idx] = mi | ((mv > 0.f) ? (1 << 30) : 0);
}

__global__ void inhib_a2(const int* __restrict__ pa, int* __restrict__ winfo) {
    int hw = blockIdx.x * 256 + threadIdx.x;
    if (hw >= HW2) return;
    unsigned long long amlo = 0, amhi = 0;
    unsigned cm = 0;
#pragma unroll
    for (int t = 0; t < T; ++t) {
        int p = pa[t * HW2 + hw];
        int mi = p & 0xff;
        if (p >> 30) cm |= 1u << t;
        if (t < 8) amlo |= (unsigned long long)mi << (8 * t);
        else       amhi |= (unsigned long long)mi << (8 * (t - 8));
    }
    int ns = __popc(cm);
    int e = 15 - ns; e = e < 0 ? 0 : (e > 14 ? 14 : e);
    int winner = (int)((e < 8 ? (amlo >> (8 * e)) : (amhi >> (8 * (e - 8)))) & 255ull);
    int spiked = (cm >> 14) & 1;
    winfo[hw] = winner | (spiked << 16);
}

__global__ void inhib_b(float* __restrict__ pot, float* __restrict__ spk,
                        const int* __restrict__ winfo) {
    long idx = (long)blockIdx.x * 256 + threadIdx.x;          // (t,c,h,w)
    if (idx >= NOUT) return;
    int hw = (int)(idx & (HW2 - 1));
    int cc = (int)((idx >> 14) % C2O);
    int wi = winfo[hw];
    int winner = wi & 0xffff, spiked = wi >> 16;
    float v = pot[idx];
    float o = (spiked && cc == winner) ? v : 0.f;
    pot[idx] = o;
    spk[idx] = (o > 0.f) ? 1.f : 0.f;
}

// ---------------- k-winners ----------------
__global__ void kw_init(unsigned* vmax) { *vmax = 0u; }

__global__ void kw_prep(const float* __restrict__ pot, float* __restrict__ nsp,
                        float* __restrict__ val, unsigned* __restrict__ vmax) {
    __shared__ float sm[256];
    int idx = blockIdx.x * 256 + threadIdx.x;                 // (c,h,w)
    float tr = 0.f;
    if (idx < NTOT) {
        int ns = 0;
#pragma unroll
        for (int t = 0; t < T; ++t) ns += (pot[(size_t)t * NTOT + idx] > 0.f) ? 1 : 0;
        int e = 15 - ns; e = e < 0 ? 0 : (e > 14 ? 14 : e);
        float v = pot[(size_t)e * NTOT + idx];
        nsp[idx] = (float)ns;
        val[idx] = v;
        tr = ns > 0 ? v : 0.f;
    }
    sm[threadIdx.x] = tr;
    __syncthreads();
    for (int s = 128; s > 0; s >>= 1) {
        if (threadIdx.x < s) sm[threadIdx.x] = fmaxf(sm[threadIdx.x], sm[threadIdx.x + s]);
        __syncthreads();
    }
    if (threadIdx.x == 0) atomicMax(vmax, __float_as_uint(sm[0]));
}

__global__ void kw_total(const float* __restrict__ nsp, const float* __restrict__ val,
                         const unsigned* __restrict__ vmax, float* __restrict__ total) {
    int idx = blockIdx.x * 256 + threadIdx.x;
    if (idx >= NTOT) return;
    float v = __uint_as_float(*vmax) * (float)T;
    total[idx] = nsp[idx] * (val[idx] + v);
}

__global__ void kw_r1(const float* __restrict__ total, float* __restrict__ redv,
                      int* __restrict__ redi) {
    __shared__ float sv[256];
    __shared__ int   si[256];
    float bv = -1.f; int bi = 0x7fffffff;
    for (int i = blockIdx.x * 256 + threadIdx.x; i < NTOT; i += 1024 * 256) {
        float v = total[i];
        if (v > bv) { bv = v; bi = i; }
    }
    sv[threadIdx.x] = bv; si[threadIdx.x] = bi;
    __syncthreads();
    for (int s = 128; s > 0; s >>= 1) {
        if (threadIdx.x < s) {
            if (sv[threadIdx.x + s] > sv[threadIdx.x] ||
                (sv[threadIdx.x + s] == sv[threadIdx.x] && si[threadIdx.x + s] < si[threadIdx.x])) {
                sv[threadIdx.x] = sv[threadIdx.x + s];
                si[threadIdx.x] = si[threadIdx.x + s];
            }
        }
        __syncthreads();
    }
    if (threadIdx.x == 0) { redv[blockIdx.x] = sv[0]; redi[blockIdx.x] = si[0]; }
}

__global__ void kw_r2(const float* __restrict__ redv, const int* __restrict__ redi,
                      int* __restrict__ wb, float* __restrict__ winners, int slot) {
    __shared__ float sv[256];
    __shared__ int   si[256];
    float bv = -1.f; int bi = 0x7fffffff;
    for (int j = threadIdx.x; j < 1024; j += 256) {
        float v = redv[j]; int ii = redi[j];
        if (v > bv || (v == bv && ii < bi)) { bv = v; bi = ii; }
    }
    sv[threadIdx.x] = bv; si[threadIdx.x] = bi;
    __syncthreads();
    for (int s = 128; s > 0; s >>= 1) {
        if (threadIdx.x < s) {
            if (sv[threadIdx.x + s] > sv[threadIdx.x] ||
                (sv[threadIdx.x + s] == sv[threadIdx.x] && si[threadIdx.x + s] < si[threadIdx.x])) {
                sv[threadIdx.x] = sv[threadIdx.x + s];
                si[threadIdx.x] = si[threadIdx.x + s];
            }
        }
        __syncthreads();
    }
    if (threadIdx.x == 0) {
        int idx = si[0];
        int c = idx >> 14, rr = (idx >> 7) & 127, cc = idx & 127;
        wb[0] = c; wb[1] = rr; wb[2] = cc;                    // kill uses argmax pos even if max==0
        bool valid = sv[0] != 0.f;
        winners[slot * 3 + 0] = valid ? (float)c  : -1.f;
        winners[slot * 3 + 1] = valid ? (float)rr : -1.f;
        winners[slot * 3 + 2] = valid ? (float)cc : -1.f;
    }
}

__global__ void kw_sup(float* __restrict__ total, const int* __restrict__ wb) {
    int idx = blockIdx.x * 256 + threadIdx.x;
    if (idx >= NTOT) return;
    int c = idx >> 14, h = (idx >> 7) & 127, w = idx & 127;
    int wc = wb[0], wr = wb[1], wcc = wb[2];
    bool kill = (c == wc) || (abs(h - wr) <= 1 && abs(w - wcc) <= 1);
    if (kill) total[idx] = 0.f;
}

// ---------------- driver ----------------
extern "C" void kernel_launch(void* const* d_in, const int* in_sizes, int n_in,
                              void* d_out, int out_size, void* d_ws, size_t ws_size,
                              hipStream_t stream) {
    (void)in_sizes; (void)n_in; (void)out_size; (void)ws_size;
    const float* x  = (const float*)d_in[0];
    const float* w1 = (const float*)d_in[1];
    const float* w2 = (const float*)d_in[2];

    char* ws = (char*)d_ws;
    _Float16* w1p  = (_Float16*)(ws + W1P_OFF);
    _Float16* w2p  = (_Float16*)(ws + W2P_OFF);
    _Float16* xpad = (_Float16*)(ws + XPAD_OFF);
    _Float16* ppad = (_Float16*)(ws + PPAD_OFF);
    int*      pa   = (int*)     (ws + PA_OFF);
    int*      winfo= (int*)     (ws + WIN_OFF);
    float*    nsp  = (float*)   (ws + NSP_OFF);
    float*    val  = (float*)   (ws + VAL_OFF);
    float*    tot  = (float*)   (ws + TOT_OFF);
    unsigned* vmax = (unsigned*)(ws + VMAX_OFF);
    float*    redv = (float*)   (ws + REDV_OFF);
    int*      redi = (int*)     (ws + REDI_OFF);
    int*      wb   = (int*)     (ws + WB_OFF);

    float* spk_out = (float*)d_out;
    float* pot_out = (float*)d_out + NOUT;
    float* win_out = (float*)d_out + 2 * NOUT;
    _Float16* spk1 = (_Float16*)d_out;                        // conv1 spikes parked in spk region

    kw_init<<<1, 1, 0, stream>>>(vmax);
    prep_w1<<<(KT1 * NT1 * 32 * 16 + 255) / 256, 256, 0, stream>>>(w1, w1p);
    prep_w2<<<(KT2 * NT2 * 32 * 16 + 255) / 256, 256, 0, stream>>>(w2, w2p);
    xpad_fill<<<(T * C1I * HP1 * HP1 + 255) / 256, 256, 0, stream>>>(x, xpad);

    conv1_fire<<<7680, 256, 0, stream>>>(xpad, w1p, spk1);    // 61,440 tiles / 8 waves
    pool_k<<<(T * C1O * HP2 * HP2 + 255) / 256, 256, 0, stream>>>(spk1, ppad);
    conv2_fire<<<1920, 256, 0, stream>>>(ppad, w2p, pot_out); // 15,360 tiles / 8 waves

    inhib_a1<<<(T * HW2) / 256, 256, 0, stream>>>(pot_out, pa);
    inhib_a2<<<HW2 / 256, 256, 0, stream>>>(pa, winfo);
    inhib_b<<<(int)(NOUT / 256), 256, 0, stream>>>(pot_out, spk_out, winfo);

    kw_prep<<<NTOT / 256, 256, 0, stream>>>(pot_out, nsp, val, vmax);
    kw_total<<<NTOT / 256, 256, 0, stream>>>(nsp, val, vmax, tot);

    for (int i = 0; i < 8; ++i) {
        kw_r1<<<1024, 256, 0, stream>>>(tot, redv, redi);
        kw_r2<<<1, 256, 0, stream>>>(redv, redi, wb, win_out, i);
        kw_sup<<<NTOT / 256, 256, 0, stream>>>(tot, wb);
    }
}
